// AIGMAEEncoder_69930657513567
// MI455X (gfx1250) — matrix operand order repaired and checked by reading, not compiled, not run on hardware
//
#include <hip/hip_runtime.h>
#include <cstdint>

#define Gg   8
#define Nn   8192
#define Ee   16384
#define Hh   256
#define H2   512
#define Ll   4
#define EPSF 1e-7f

typedef __attribute__((ext_vector_type(16))) __bf16 v16bf;
typedef __attribute__((ext_vector_type(8)))  __bf16 v8bf;
typedef __attribute__((ext_vector_type(8)))  float  v8f;

// ---------------------------------------------------------------------------
// CDNA5 async global->LDS copy (GVS mode), tracked by ASYNCcnt.
// ---------------------------------------------------------------------------
__device__ __forceinline__ void async_copy_b128(unsigned lds_off, unsigned goff,
                                                const void* base) {
    asm volatile("global_load_async_to_lds_b128 %0, %1, %2"
                 :
                 : "v"(lds_off), "v"(goff),
                   "s"((unsigned long long)(uintptr_t)base)
                 : "memory");
}

__device__ __forceinline__ void wait_async0() {
    asm volatile("s_wait_asynccnt 0x0" ::: "memory");
}
__device__ __forceinline__ void wait_async4() {
    asm volatile("s_wait_asynccnt 0x4" ::: "memory");
}

// issue one 16 KB stage: 1024 x B128, 4 per thread (256 threads)
__device__ __forceinline__ void stage16k(unsigned ldsBase, const void* gsrc, int tid) {
    #pragma unroll
    for (int j = 0; j < 4; ++j) {
        unsigned i16 = (unsigned)(tid + j * 256) * 16u;
        async_copy_b128(ldsBase + i16, i16, gsrc);
    }
}

__device__ __forceinline__ v16bf catbf(v8bf lo, v8bf hi) {
    v16bf r;
    #pragma unroll
    for (int i = 0; i < 8; ++i) { r[i] = lo[i]; r[i + 8] = hi[i]; }
    return r;
}

// ---------------------------------------------------------------------------
// float atomic max via signed/unsigned int ordering trick
// ---------------------------------------------------------------------------
__device__ __forceinline__ void atomicMaxFloat(float* addr, float v) {
    if (v >= 0.0f) {
        atomicMax((int*)addr, __float_as_int(v));
    } else {
        atomicMin((unsigned int*)addr, (unsigned int)__float_as_int(v));
    }
}

__device__ __forceinline__ float fixFinite(float v) {
    if (!(fabsf(v) <= 3.0e38f)) return 0.0f;
    return v;
}

// ---------------------------------------------------------------------------
// initial residual-stream copy: out = input_nodes
// ---------------------------------------------------------------------------
__global__ void copy_kernel(const float4* __restrict__ in, float4* __restrict__ out, long n4) {
    long i = (long)blockIdx.x * blockDim.x + threadIdx.x;
    if (i < n4) out[i] = in[i];
}

// ---------------------------------------------------------------------------
// Weight pre-swizzle into WMMA B-fragment order (bf16), 16 KB stages.
// ---------------------------------------------------------------------------
__global__ void swz_w1_kernel(const float* __restrict__ W1, __bf16* __restrict__ out) {
    long i = (long)blockIdx.x * blockDim.x + threadIdx.x;   // Ll*Hh*H2
    int layer = (int)(i / (Hh * H2));
    int r = (int)(i % (Hh * H2));
    int s = r >> 13;            // 16 KB stage = 8192 bf16
    int q = r & 8191;
    int tile = q >> 9;          // 0..15
    int lane = (q >> 4) & 31;
    int e = q & 15;
    int kc = s >> 1, half = s & 1;
    int n = half * 256 + tile * 16 + (lane & 15);
    int k = kc * 32 + (lane >> 4) * 8 + (e & 7) + (e >> 3) * 16;
    out[i] = (__bf16)W1[(long)layer * Hh * H2 + (long)k * H2 + n];
}

__global__ void swz_w2_kernel(const float* __restrict__ W2, __bf16* __restrict__ out) {
    long i = (long)blockIdx.x * blockDim.x + threadIdx.x;   // Ll*H2*Hh
    int layer = (int)(i / (H2 * Hh));
    int r = (int)(i % (H2 * Hh));
    int s = r >> 13;
    int q = r & 8191;
    int tile = q >> 9;
    int lane = (q >> 4) & 31;
    int e = q & 15;
    int n = tile * 16 + (lane & 15);
    int k = s * 32 + (lane >> 4) * 8 + (e & 7) + (e >> 3) * 16;
    out[i] = (__bf16)W2[(long)layer * H2 * Hh + (long)k * Hh + n];
}

// ---------------------------------------------------------------------------
// LayerNorm of x into h; init mx=-inf, den=0, agg=0.  One wave per node.
// ---------------------------------------------------------------------------
__global__ void ln_init_kernel(const float* __restrict__ x,
                               const float* __restrict__ ln_g,
                               const float* __restrict__ ln_b,
                               float* __restrict__ h,
                               float* __restrict__ mx,
                               float* __restrict__ den,
                               float* __restrict__ agg,
                               int layer) {
    const int wave = threadIdx.x >> 5;
    const int lane = threadIdx.x & 31;
    const long node = (long)blockIdx.x * 8 + wave;

    const float4* xr = (const float4*)(x + node * Hh);
    float4 v0 = xr[lane];
    float4 v1 = xr[lane + 32];

    float s  = v0.x + v0.y + v0.z + v0.w + v1.x + v1.y + v1.z + v1.w;
    float ss = v0.x*v0.x + v0.y*v0.y + v0.z*v0.z + v0.w*v0.w
             + v1.x*v1.x + v1.y*v1.y + v1.z*v1.z + v1.w*v1.w;
    #pragma unroll
    for (int m = 16; m >= 1; m >>= 1) {
        s  += __shfl_xor(s,  m, 32);
        ss += __shfl_xor(ss, m, 32);
    }
    const float mu  = s * (1.0f / Hh);
    const float var = ss * (1.0f / Hh) - mu * mu;
    const float rs  = rsqrtf(var + 1e-5f);

    const float4* gp = (const float4*)(ln_g + (long)layer * Hh);
    const float4* bp = (const float4*)(ln_b + (long)layer * Hh);
    float4 g0 = gp[lane], g1 = gp[lane + 32];
    float4 b0 = bp[lane], b1 = bp[lane + 32];

    float4 h0, h1;
    h0.x = (v0.x - mu) * rs * g0.x + b0.x;  h0.y = (v0.y - mu) * rs * g0.y + b0.y;
    h0.z = (v0.z - mu) * rs * g0.z + b0.z;  h0.w = (v0.w - mu) * rs * g0.w + b0.w;
    h1.x = (v1.x - mu) * rs * g1.x + b1.x;  h1.y = (v1.y - mu) * rs * g1.y + b1.y;
    h1.z = (v1.z - mu) * rs * g1.z + b1.z;  h1.w = (v1.w - mu) * rs * g1.w + b1.w;

    float4* hr = (float4*)(h + node * Hh);
    hr[lane] = h0; hr[lane + 32] = h1;

    const float ninf = -__builtin_inff();
    float4 vinf = make_float4(ninf, ninf, ninf, ninf);
    float4 vz   = make_float4(0.f, 0.f, 0.f, 0.f);
    float4* mr = (float4*)(mx  + node * Hh);
    float4* dr = (float4*)(den + node * Hh);
    float4* ar = (float4*)(agg + node * Hh);
    mr[lane] = vinf; mr[lane + 32] = vinf;
    dr[lane] = vz;   dr[lane + 32] = vz;
    ar[lane] = vz;   ar[lane + 32] = vz;
}

// ---------------------------------------------------------------------------
// Edge pass A: segment max of logits into mx
// ---------------------------------------------------------------------------
__global__ void edge_max_kernel(const int* __restrict__ edges,
                                const float* __restrict__ h,
                                const float* __restrict__ t,
                                float* __restrict__ mx,
                                int layer) {
    long idx = (long)blockIdx.x * blockDim.x + threadIdx.x;  // Gg*Ee*(Hh/4)
    const int c4 = (int)(idx & 63);
    long rem = idx >> 6;
    const int e = (int)(rem % Ee);
    const int g = (int)(rem / Ee);
    const int src = edges[(long)g * 2 * Ee + e];
    const int dst = edges[(long)g * 2 * Ee + Ee + e];
    const float tv = t[layer];

    float4 hv = *(const float4*)(h + ((long)g * Nn + src) * Hh + c4 * 4);
    float* mp = mx + ((long)g * Nn + dst) * Hh + c4 * 4;
    atomicMaxFloat(mp + 0, (fmaxf(hv.x, 0.f) + EPSF) * tv);
    atomicMaxFloat(mp + 1, (fmaxf(hv.y, 0.f) + EPSF) * tv);
    atomicMaxFloat(mp + 2, (fmaxf(hv.z, 0.f) + EPSF) * tv);
    atomicMaxFloat(mp + 3, (fmaxf(hv.w, 0.f) + EPSF) * tv);
}

// ---------------------------------------------------------------------------
// Edge pass B: den = segment_sum(exp(logit - mx[dst]))
// ---------------------------------------------------------------------------
__global__ void edge_sum_kernel(const int* __restrict__ edges,
                                const float* __restrict__ h,
                                const float* __restrict__ t,
                                const float* __restrict__ mx,
                                float* __restrict__ den,
                                int layer) {
    long idx = (long)blockIdx.x * blockDim.x + threadIdx.x;
    const int c4 = (int)(idx & 63);
    long rem = idx >> 6;
    const int e = (int)(rem % Ee);
    const int g = (int)(rem / Ee);
    const int src = edges[(long)g * 2 * Ee + e];
    const int dst = edges[(long)g * 2 * Ee + Ee + e];
    const float tv = t[layer];

    float4 hv = *(const float4*)(h  + ((long)g * Nn + src) * Hh + c4 * 4);
    float4 mv = *(const float4*)(mx + ((long)g * Nn + dst) * Hh + c4 * 4);
    float* dp = den + ((long)g * Nn + dst) * Hh + c4 * 4;
    atomicAdd(dp + 0, __expf((fmaxf(hv.x, 0.f) + EPSF) * tv - fixFinite(mv.x)));
    atomicAdd(dp + 1, __expf((fmaxf(hv.y, 0.f) + EPSF) * tv - fixFinite(mv.y)));
    atomicAdd(dp + 2, __expf((fmaxf(hv.z, 0.f) + EPSF) * tv - fixFinite(mv.z)));
    atomicAdd(dp + 3, __expf((fmaxf(hv.w, 0.f) + EPSF) * tv - fixFinite(mv.w)));
}

// ---------------------------------------------------------------------------
// Edge pass C: agg = segment_sum(m * ex / max(den[dst], eps))
// ---------------------------------------------------------------------------
__global__ void edge_agg_kernel(const int* __restrict__ edges,
                                const float* __restrict__ h,
                                const float* __restrict__ t,
                                const float* __restrict__ mx,
                                const float* __restrict__ den,
                                float* __restrict__ agg,
                                int layer) {
    long idx = (long)blockIdx.x * blockDim.x + threadIdx.x;
    const int c4 = (int)(idx & 63);
    long rem = idx >> 6;
    const int e = (int)(rem % Ee);
    const int g = (int)(rem / Ee);
    const int src = edges[(long)g * 2 * Ee + e];
    const int dst = edges[(long)g * 2 * Ee + Ee + e];
    const float tv = t[layer];

    float4 hv = *(const float4*)(h   + ((long)g * Nn + src) * Hh + c4 * 4);
    float4 mv = *(const float4*)(mx  + ((long)g * Nn + dst) * Hh + c4 * 4);
    float4 dv = *(const float4*)(den + ((long)g * Nn + dst) * Hh + c4 * 4);
    float* ap = agg + ((long)g * Nn + dst) * Hh + c4 * 4;

    float m0 = fmaxf(hv.x, 0.f) + EPSF, m1 = fmaxf(hv.y, 0.f) + EPSF;
    float m2 = fmaxf(hv.z, 0.f) + EPSF, m3 = fmaxf(hv.w, 0.f) + EPSF;
    atomicAdd(ap + 0, m0 * __expf(m0 * tv - fixFinite(mv.x)) / fmaxf(dv.x, EPSF));
    atomicAdd(ap + 1, m1 * __expf(m1 * tv - fixFinite(mv.y)) / fmaxf(dv.y, EPSF));
    atomicAdd(ap + 2, m2 * __expf(m2 * tv - fixFinite(mv.z)) / fmaxf(dv.z, EPSF));
    atomicAdd(ap + 3, m3 * __expf(m3 * tv - fixFinite(mv.w)) / fmaxf(dv.w, EPSF));
}

// ---------------------------------------------------------------------------
// WMMA MLP with double-buffered async weight staging.
// One block = 16 nodes, 256 threads = 8 wave32s.
// 16 stages of 16 KB per GEMM; stage s -> LDS buf (s&1); prefetch s+2 after use.
// Accumulators are named scalars so WMMA dst/src are fixed VGPR ranges
// (no v_movrels/v_movreld indexed copies).
// ---------------------------------------------------------------------------
__global__ void mlp_kernel(const float* x,          // residual stream (== out)
                           const float* __restrict__ h,
                           const float* __restrict__ agg,
                           const __bf16* __restrict__ W1s, const float* __restrict__ b1,
                           const float* __restrict__ mg, const float* __restrict__ mb,
                           const __bf16* __restrict__ W2s, const float* __restrict__ b2,
                           const float* __restrict__ mask,
                           float* out,
                           int layer) {
    __shared__ __align__(16) unsigned char smem[57344];
    __bf16* Abuf  = (__bf16*)smem;              // 16 x 256 bf16      (8 KB)
    __bf16* A1buf = (__bf16*)(smem + 8192);     // 16 x 512 bf16      (16 KB)
    __bf16* Wb    = (__bf16*)(smem + 24576);    // 2 x 16 KB ping-pong (32 KB)
    float*  Y1    = (float*)(smem + 24576);     // 16 x 512 f32 (aliases Wb)

    const int tid    = threadIdx.x;
    const int wave   = tid >> 5;
    const int lane   = tid & 31;
    const int laneHi = lane >> 4;
    const int ln16   = lane & 15;
    const long nodeBase = (long)blockIdx.x * 16;
    const unsigned WbOff = (unsigned)(uintptr_t)(void*)Wb;

    const __bf16* W1l = W1s + (long)layer * Hh * H2;   // stage-major swizzled
    const float*  b1l = b1 + (long)layer * H2;
    const float*  mgl = mg + (long)layer * H2;
    const float*  mbl = mb + (long)layer * H2;
    const __bf16* W2l = W2s + (long)layer * H2 * Hh;   // stage-major swizzled
    const float*  b2l = b2 + (long)layer * Hh;

    // ---- stage z tile (16 x 256) as bf16 ----
    for (int i = tid; i < 16 * Hh; i += 256) {
        int m = i >> 8, c = i & 255;
        long off = (nodeBase + m) * (long)Hh + c;
        Abuf[i] = (__bf16)(agg[off] + h[off]);
    }
    // prologue: stages 0,1 in flight
    stage16k(WbOff,         W1l,        tid);
    stage16k(WbOff + 16384, W1l + 8192, tid);
    __syncthreads();   // Abuf visible before first A-fragment read

    // ---- GEMM1: (16x256)@(256x512); stage s = {kc, half} ----
    v8f a10 = {}, a11 = {}, a12 = {}, a13 = {};
    for (int kc = 0; kc < 8; ++kc) {
        const __bf16* ar = Abuf + ln16 * Hh + kc * 32 + laneHi * 8;
        v16bf a1 = catbf(*(const v8bf*)ar, *(const v8bf*)(ar + 16));
        #pragma unroll
        for (int half = 0; half < 2; ++half) {
            const int s = kc * 2 + half;
            if (s < 15) wait_async4(); else wait_async0();
            __syncthreads();
            const __bf16* Wcur = Wb + (s & 1) * 8192;
            const __bf16* bp0 = Wcur + (2 * wave) * 512 + lane * 16;
            v16bf b0 = catbf(*(const v8bf*)bp0, *(const v8bf*)(bp0 + 8));
            v16bf b1v = catbf(*(const v8bf*)(bp0 + 512), *(const v8bf*)(bp0 + 520));
            if (half == 0) {
                a10 = __builtin_amdgcn_wmma_f32_16x16x32_bf16(
                    false, a1, false, b0, (short)0, a10, false, false);
                a11 = __builtin_amdgcn_wmma_f32_16x16x32_bf16(
                    false, a1, false, b1v, (short)0, a11, false, false);
            } else {
                a12 = __builtin_amdgcn_wmma_f32_16x16x32_bf16(
                    false, a1, false, b0, (short)0, a12, false, false);
                a13 = __builtin_amdgcn_wmma_f32_16x16x32_bf16(
                    false, a1, false, b1v, (short)0, a13, false, false);
            }
            __syncthreads();
            if (s + 2 < 16)
                stage16k(WbOff + (s & 1) * 16384, W1l + (long)(s + 2) * 8192, tid);
        }
    }

    // ---- spill y1 to LDS (Wb region now free) ----
    {
        int n00 = (2 * wave) * 16;
        #pragma unroll
        for (int r = 0; r < 8; ++r) {
            int row = r + (laneHi ? 8 : 0);
            Y1[row * H2 + n00 + ln16]            = a10[r];
            Y1[row * H2 + n00 + 16 + ln16]       = a11[r];
            Y1[row * H2 + 256 + n00 + ln16]      = a12[r];
            Y1[row * H2 + 256 + n00 + 16 + ln16] = a13[r];
        }
    }
    __syncthreads();

    // ---- row LayerNorm over 512 + ReLU, wave w does rows w and w+8 ----
    for (int rr = 0; rr < 2; ++rr) {
        int row = wave + rr * 8;
        float s = 0.f, ss = 0.f, vals[16];
        #pragma unroll
        for (int j = 0; j < 16; ++j) {
            int col = lane * 16 + j;
            float v = Y1[row * H2 + col] + b1l[col];
            vals[j] = v; s += v; ss += v * v;
        }
        #pragma unroll
        for (int mk = 16; mk >= 1; mk >>= 1) {
            s  += __shfl_xor(s,  mk, 32);
            ss += __shfl_xor(ss, mk, 32);
        }
        float mu  = s * (1.0f / H2);
        float var = ss * (1.0f / H2) - mu * mu;
        float rsv = rsqrtf(var + 1e-5f);
        #pragma unroll
        for (int j = 0; j < 16; ++j) {
            int col = lane * 16 + j;
            float v = (vals[j] - mu) * rsv * mgl[col] + mbl[col];
            A1buf[row * H2 + col] = (__bf16)fmaxf(v, 0.f);
        }
    }
    __syncthreads();   // Y1 dead, Wb region reusable

    // prologue for GEMM2
    stage16k(WbOff,         W2l,        tid);
    stage16k(WbOff + 16384, W2l + 8192, tid);

    // ---- GEMM2: (16x512)@(512x256); stage s = kc ----
    v8f a20 = {}, a21 = {};
    for (int s = 0; s < 16; ++s) {
        if (s < 15) wait_async4(); else wait_async0();
        __syncthreads();
        const __bf16* ar = A1buf + ln16 * H2 + s * 32 + laneHi * 8;
        v16bf a2 = catbf(*(const v8bf*)ar, *(const v8bf*)(ar + 16));
        const __bf16* Wcur = Wb + (s & 1) * 8192;
        const __bf16* bp0 = Wcur + (2 * wave) * 512 + lane * 16;
        v16bf b0 = catbf(*(const v8bf*)bp0, *(const v8bf*)(bp0 + 8));
        v16bf b1v = catbf(*(const v8bf*)(bp0 + 512), *(const v8bf*)(bp0 + 520));
        a20 = __builtin_amdgcn_wmma_f32_16x16x32_bf16(
            false, a2, false, b0, (short)0, a20, false, false);
        a21 = __builtin_amdgcn_wmma_f32_16x16x32_bf16(
            false, a2, false, b1v, (short)0, a21, false, false);
        __syncthreads();
        if (s + 2 < 16)
            stage16k(WbOff + (s & 1) * 16384, W2l + (long)(s + 2) * 8192, tid);
    }

    // ---- epilogue: +b2, residual, mask, inter-layer ReLU, write out ----
    #pragma unroll
    for (int j = 0; j < 2; ++j) {
        int n0 = (2 * wave + j) * 16;
        #pragma unroll
        for (int r = 0; r < 8; ++r) {
            int row = r + (laneHi ? 8 : 0);
            long node = nodeBase + row;
            int g = (int)(node / Nn);
            long ng = node % Nn;
            int col = n0 + ln16;
            long off = node * (long)Hh + col;
            float acc = (j == 0) ? a20[r] : a21[r];
            float v = acc + b2l[col] + x[off];
            v *= mask[(long)g * Nn + ng];
            if (layer < Ll - 1) v = fmaxf(v, 0.f);
            out[off] = v;
        }
    }
}

// ---------------------------------------------------------------------------
extern "C" void kernel_launch(void* const* d_in, const int* in_sizes, int n_in,
                              void* d_out, int out_size, void* d_ws, size_t ws_size,
                              hipStream_t stream) {
    const float* input_nodes = (const float*)d_in[0];
    const int*   edges       = (const int*)d_in[1];
    const float* mask        = (const float*)d_in[2];
    const float* ln_g        = (const float*)d_in[3];
    const float* ln_b        = (const float*)d_in[4];
    const float* t           = (const float*)d_in[5];
    const float* W1          = (const float*)d_in[6];
    const float* b1          = (const float*)d_in[7];
    const float* mg          = (const float*)d_in[8];
    const float* mb          = (const float*)d_in[9];
    const float* W2          = (const float*)d_in[10];
    const float* b2          = (const float*)d_in[11];
    float* out = (float*)d_out;

    const size_t tensorElems = (size_t)Gg * Nn * Hh;   // 16.7M floats
    float* h   = (float*)d_ws;
    float* mx  = h   + tensorElems;
    float* den = mx  + tensorElems;
    float* agg = den + tensorElems;
    __bf16* W1sw = (__bf16*)(agg + tensorElems);
    __bf16* W2sw = W1sw + (size_t)Ll * Hh * H2;

    const long n4 = (long)(tensorElems / 4);
    copy_kernel<<<(int)((n4 + 255) / 256), 256, 0, stream>>>(
        (const float4*)input_nodes, (float4*)out, n4);

    const long nW = (long)Ll * Hh * H2;   // 524288 per weight tensor
    swz_w1_kernel<<<(int)(nW / 256), 256, 0, stream>>>(W1, W1sw);
    swz_w2_kernel<<<(int)(nW / 256), 256, 0, stream>>>(W2, W2sw);

    const int edgeBlocks = (int)(((long)Gg * Ee * (Hh / 4)) / 256);  // 32768
    const int nodeBlocks = (int)(((long)Gg * Nn) / 8);               // 8192
    const int tileBlocks = (int)(((long)Gg * Nn) / 16);              // 4096

    for (int layer = 0; layer < Ll; ++layer) {
        ln_init_kernel<<<nodeBlocks, 256, 0, stream>>>(out, ln_g, ln_b,
                                                       h, mx, den, agg, layer);
        edge_max_kernel<<<edgeBlocks, 256, 0, stream>>>(edges, h, t, mx, layer);
        edge_sum_kernel<<<edgeBlocks, 256, 0, stream>>>(edges, h, t, mx, den, layer);
        edge_agg_kernel<<<edgeBlocks, 256, 0, stream>>>(edges, h, t, mx, den, agg, layer);
        mlp_kernel<<<tileBlocks, 256, 0, stream>>>(out, h, agg, W1sw, b1, mg, mb,
                                                   W2sw, b2, mask, out, layer);
    }
}
